// SparseConv2d_39582418600033
// MI455X (gfx1250) — compile-verified, matrix operand order checked by reading, not run on hardware
//
#include <hip/hip_runtime.h>
#include <hip/hip_bf16.h>
#include <stdint.h>

typedef __attribute__((ext_vector_type(8)))  _Float16 v8h;
typedef __attribute__((ext_vector_type(16))) _Float16 v16h;
typedef __attribute__((ext_vector_type(8)))  float    v8f;

// Problem dims
constexpr int BATCH = 32, CIN = 256, H = 56, W = 56, OCH = 256, JC = 64;
constexpr int HW   = H * W;          // 3136
constexpr int KTOT = 9 * CIN;        // 2304 dense-K (tap-major, channel-minor)

// Tiling
constexpr int TM = 128, TN = 64;
constexpr int LDA = 40;              // halves (32+8 pad)
constexpr int LDB = 264;             // halves (256+8 pad)
constexpr int LDC = 132;             // floats (128+4 pad)
constexpr int NSTEP   = 72;          // 9 taps x 8 channel-blocks of 32
constexpr int AT_BUF  = TM * LDA * 2;        // 10240 B per A buffer
constexpr int BT_OFF  = 3 * AT_BUF;          // 30720
constexpr int SMEM_SZ = BT_OFF + TN * LDB * 2; // 64512 B
constexpr int MTILES = BATCH * HW / TM;  // 784
constexpr int NTILES = OCH / TN;         // 4

// ---------------------------------------------------------------------------
// 1) NCHW fp32 -> NHWC f16 transpose (one pass, result is L2-resident, 51 MB)
// ---------------------------------------------------------------------------
__global__ __launch_bounds__(256) void nchw_to_nhwc_f16(
    const float* __restrict__ x, _Float16* __restrict__ xh) {
  __shared__ _Float16 tbuf[CIN * 65];
  const int tid = threadIdx.x;
  const int sp0 = blockIdx.x * 64;     // 3136 % 64 == 0: never crosses batch
  const int b   = sp0 / HW;
  const int hw0 = sp0 % HW;

  for (int it = 0; it < 16; ++it) {    // coalesced float4 reads along hw
    const int c  = it * 16 + (tid >> 4);
    const int hp = (tid & 15) * 4;
    const float4 v = *(const float4*)(x + ((size_t)(b * CIN + c)) * HW + hw0 + hp);
    tbuf[c * 65 + hp + 0] = (_Float16)v.x;
    tbuf[c * 65 + hp + 1] = (_Float16)v.y;
    tbuf[c * 65 + hp + 2] = (_Float16)v.z;
    tbuf[c * 65 + hp + 3] = (_Float16)v.w;
  }
  __syncthreads();

  const int sl = tid >> 2;             // coalesced b128 writes along c
  const int cp = (tid & 3) * 64;
  _Float16* dst = xh + (size_t)(sp0 + sl) * CIN + cp;
#pragma unroll
  for (int u = 0; u < 8; ++u) {
    v8h t;
#pragma unroll
    for (int j = 0; j < 8; ++j) t[j] = tbuf[(cp + u * 8 + j) * 65 + sl];
    *(v8h*)(dst + u * 8) = t;
  }
}

// ---------------------------------------------------------------------------
// 2) zero dense-weight buffer, 3) scatter sparse weights -> wd[o][tap*256+c]
// ---------------------------------------------------------------------------
__global__ void zero_u32(uint32_t* __restrict__ p, int n) {
  const int i = blockIdx.x * 256 + threadIdx.x;
  if (i < n) p[i] = 0u;
}

__global__ void scatter_w(const float* __restrict__ wv, const int* __restrict__ wi,
                          _Float16* __restrict__ wd) {
  const int g = blockIdx.x * 256 + threadIdx.x;  // O*J threads
  const int o = g >> 6;
  const int j = g & 63;
  const int c = wi[o * JC + j];
  const float* src = wv + (size_t)(o * JC + j) * 9;
#pragma unroll
  for (int t = 0; t < 9; ++t)
    wd[(size_t)o * KTOT + t * CIN + c] = (_Float16)src[t];
}

// ---------------------------------------------------------------------------
// 4) implicit-GEMM conv, async-to-LDS pipelined, 4 WMMAs/wave/K-step
// ---------------------------------------------------------------------------
__global__ __launch_bounds__(256) void spconv_wmma(
    const _Float16* __restrict__ xh,   // NHWC f16 activations
    const _Float16* __restrict__ wd,   // dense f16 weights [O][KTOT]
    float* __restrict__ out) {
  __shared__ __align__(16) char smem[SMEM_SZ];
  _Float16* Bt = (_Float16*)(smem + BT_OFF);
  float*    Ct = (float*)smem;         // epilogue alias over A buffers

  const int tid   = threadIdx.x;
  const int lane  = tid & 31;
  const int wid   = tid >> 5;
  const int lrow  = lane & 15;
  const int lhalf = lane >> 4;             // K chunk select (ISA 16-bit A layout)
  const int wm    = (wid & 3) * 32;        // wave 32(M) x 32(N) tile
  const int wn    = (wid >> 2) * 32;
  const int m0    = blockIdx.x * TM;
  const int n0    = blockIdx.y * TN;

  // A staging: thread owns spatial pos ms, 16 channels (cpart half of 32-block)
  const int ms    = tid & (TM - 1);
  const int cpart = tid >> 7;
  const int mg    = m0 + ms;
  const int bi    = mg / HW;
  const int hwm   = mg % HW;
  const int hh    = hwm / W;
  const int ww    = hwm % W;

  // global A source for K-step s (address clamped when in the zero-pad border)
  auto a_src = [&](int s, bool& vld) -> const _Float16* {
    const int t = s >> 3, c0 = (s & 7) * 32;
    const int y = hh + t / 3 - 1, xc = ww + (t % 3) - 1;
    vld = (y >= 0) & (y < H) & (xc >= 0) & (xc < W);
    const size_t sp = vld ? ((size_t)bi * HW + y * W + xc) : 0;
    return xh + sp * CIN + cpart * 16 + c0;
  };
  // LDS byte address of this thread's slot in A buffer for step s
  auto a_dst = [&](int s) -> uint32_t {
    return (uint32_t)(uintptr_t)(smem) + (uint32_t)((s % 3) * AT_BUF)
         + (uint32_t)((ms * LDA + cpart * 16) * 2);
  };

  // prologue: async prefetch of step 0 A tile
  bool vld_cur;
  {
    const _Float16* g0 = a_src(0, vld_cur);
    const uint32_t  l0 = a_dst(0);
    asm volatile(
        "global_load_async_to_lds_b128 %0, %1, off\n\t"
        "global_load_async_to_lds_b128 %0, %1, off offset:16"
        :: "v"(l0), "v"((uint64_t)(uintptr_t)g0) : "memory");
  }

  v8f acc00 = {}, acc01 = {}, acc10 = {}, acc11 = {};

  const int bnr = tid >> 2;            // B staging: 64 rows x 256 ch per tap
  const int bcb = (tid & 3) * 64;      // 64 contiguous halves / thread

  for (int s = 0; s < NSTEP; ++s) {
    const int c0 = (s & 7) * 32;

    if ((s & 7) == 0) {                // new tap: restage Bt (async, 8x b128)
      __syncthreads();                 // previous tap's Bt readers done
      const int t = s >> 3;
      const uint64_t gb = (uint64_t)(uintptr_t)(wd + (size_t)(n0 + bnr) * KTOT + t * CIN + bcb);
      const uint32_t lb = (uint32_t)(uintptr_t)(smem) + (uint32_t)BT_OFF
                        + (uint32_t)((bnr * LDB + bcb) * 2);
      asm volatile(
          "global_load_async_to_lds_b128 %0, %1, off\n\t"
          "global_load_async_to_lds_b128 %0, %1, off offset:16\n\t"
          "global_load_async_to_lds_b128 %0, %1, off offset:32\n\t"
          "global_load_async_to_lds_b128 %0, %1, off offset:48\n\t"
          "global_load_async_to_lds_b128 %0, %1, off offset:64\n\t"
          "global_load_async_to_lds_b128 %0, %1, off offset:80\n\t"
          "global_load_async_to_lds_b128 %0, %1, off offset:96\n\t"
          "global_load_async_to_lds_b128 %0, %1, off offset:112"
          :: "v"(lb), "v"(gb) : "memory");
    }

    // prefetch A for next step into buf[(s+1)%3] (clamped repeat at the tail)
    bool vld_next;
    {
      const int sn = (s + 1 < NSTEP) ? s + 1 : NSTEP - 1;
      const _Float16* gn = a_src(sn, vld_next);
      const uint32_t  ln = a_dst(s + 1);
      asm volatile(
          "global_load_async_to_lds_b128 %0, %1, off\n\t"
          "global_load_async_to_lds_b128 %0, %1, off offset:16"
          :: "v"(ln), "v"((uint64_t)(uintptr_t)gn) : "memory");
    }

    // in-order completion: <=2 outstanding  ==>  current A (and B) are in LDS
    asm volatile("s_wait_asynccnt 0x2" ::: "memory");

    // zero-fill this thread's row if it sits in the conv zero-pad border
    if (!vld_cur) {
      _Float16* z = (_Float16*)(smem + (s % 3) * AT_BUF) + ms * LDA + cpart * 16;
      v8h zz = {};
      *(v8h*)z = zz;
      *(v8h*)(z + 8) = zz;
    }
    __syncthreads();                   // single barrier per K-step
    vld_cur = vld_next;

    // per-lane fragments (two 16B chunks, +16 halves apart per ISA layout)
    const _Float16* Ab  = (const _Float16*)(smem + (s % 3) * AT_BUF);
    const _Float16* ap0 = Ab + (wm + lrow) * LDA + lhalf * 8;
    const _Float16* ap1 = ap0 + 16 * LDA;
    const v8h a0l = *(const v8h*)(ap0), a0h = *(const v8h*)(ap0 + 16);
    const v8h a1l = *(const v8h*)(ap1), a1h = *(const v8h*)(ap1 + 16);
    const _Float16* bp0 = Bt + (wn + lrow) * LDB + c0 + lhalf * 8;
    const _Float16* bp1 = bp0 + 16 * LDB;
    const v8h b0l = *(const v8h*)(bp0), b0h = *(const v8h*)(bp0 + 16);
    const v8h b1l = *(const v8h*)(bp1), b1h = *(const v8h*)(bp1 + 16);

    const v16h a0 = __builtin_shufflevector(a0l, a0h, 0,1,2,3,4,5,6,7,8,9,10,11,12,13,14,15);
    const v16h a1 = __builtin_shufflevector(a1l, a1h, 0,1,2,3,4,5,6,7,8,9,10,11,12,13,14,15);
    const v16h b0 = __builtin_shufflevector(b0l, b0h, 0,1,2,3,4,5,6,7,8,9,10,11,12,13,14,15);
    const v16h b1 = __builtin_shufflevector(b1l, b1h, 0,1,2,3,4,5,6,7,8,9,10,11,12,13,14,15);

    acc00 = __builtin_amdgcn_wmma_f32_16x16x32_f16(false, a0, false, b0, (short)0, acc00, false, false);
    acc01 = __builtin_amdgcn_wmma_f32_16x16x32_f16(false, a0, false, b1, (short)0, acc01, false, false);
    acc10 = __builtin_amdgcn_wmma_f32_16x16x32_f16(false, a1, false, b0, (short)0, acc10, false, false);
    acc11 = __builtin_amdgcn_wmma_f32_16x16x32_f16(false, a1, false, b1, (short)0, acc11, false, false);
  }

  // fence the tail dummy prefetch before aliasing A buffers as C staging
  asm volatile("s_wait_asynccnt 0x0" ::: "memory");
  __syncthreads();

  // transpose C through LDS so global stores are coalesced along W
#pragma unroll
  for (int r = 0; r < 8; ++r) {
    Ct[(wn + lrow)      * LDC + wm      + lhalf * 8 + r] = acc00[r];
    Ct[(wn + 16 + lrow) * LDC + wm      + lhalf * 8 + r] = acc01[r];
    Ct[(wn + lrow)      * LDC + wm + 16 + lhalf * 8 + r] = acc10[r];
    Ct[(wn + 16 + lrow) * LDC + wm + 16 + lhalf * 8 + r] = acc11[r];
  }
  __syncthreads();

  const int ol = tid >> 2;             // out-channel within tile (0..63)
  const int mc = (tid & 3) * 32;       // 32-aligned spatial chunk (3136%32==0)
  const int mg2 = m0 + mc;
  const int b2  = mg2 / HW;
  const int hw2 = mg2 % HW;
  float* dst = out + ((size_t)(b2 * OCH + n0 + ol)) * HW + hw2;
  const float* srow = Ct + ol * LDC + mc;
#pragma unroll
  for (int u = 0; u < 8; ++u) *(float4*)(dst + u * 4) = *(const float4*)(srow + u * 4);
}

// ---------------------------------------------------------------------------
extern "C" void kernel_launch(void* const* d_in, const int* in_sizes, int n_in,
                              void* d_out, int out_size, void* d_ws, size_t ws_size,
                              hipStream_t stream) {
  const float* x  = (const float*)d_in[0];   // [32,256,56,56] f32
  const float* wv = (const float*)d_in[1];   // [256,64,3,3]  f32
  const int*   wi = (const int*)d_in[2];     // [256,64]      i32

  _Float16* xh = (_Float16*)d_ws;                          // 51,380,224 B
  _Float16* wd = xh + (size_t)BATCH * CIN * HW;            // 1,179,648 B
  float* out = (float*)d_out;

  nchw_to_nhwc_f16<<<(BATCH * HW) / 64, 256, 0, stream>>>(x, xh);

  const int wd_u32 = OCH * KTOT / 2;                       // 294912
  zero_u32<<<(wd_u32 + 255) / 256, 256, 0, stream>>>((uint32_t*)wd, wd_u32);
  scatter_w<<<(OCH * JC) / 256, 256, 0, stream>>>(wv, wi, wd);

  spconv_wmma<<<dim3(MTILES, NTILES), 256, 0, stream>>>(xh, wd, out);
}
